// Pointnet_SA_MSG_60885456388433
// MI455X (gfx1250) — compile-verified, matrix operand order checked by reading, not run on hardware
//
#include <hip/hip_runtime.h>

typedef _Float16 v8h  __attribute__((ext_vector_type(8)));
typedef _Float16 v16h __attribute__((ext_vector_type(16)));
typedef float    v8f  __attribute__((ext_vector_type(8)));

constexpr int B_ = 8;
constexpr int N_ = 8192;
constexpr int S_ = 1024;
constexpr int C_ = 32;      // input feature channels
constexpr int FTOT = 320;   // 64 + 128 + 128 output channels
constexpr int WSTR = 104;   // padded K stride for transposed f16 weights

// ---------------------------------------------------------------------------
// Farthest point sampling: one workgroup per batch, dists in registers.
// ---------------------------------------------------------------------------
__global__ __launch_bounds__(1024) void fps_kernel(const float* __restrict__ xyz,
                                                   float* __restrict__ newXyz) {
  const int b = blockIdx.x;
  const int tid = threadIdx.x;
  const int lane = tid & 31, wid = tid >> 5;
  const float* px = xyz + (size_t)b * N_ * 3;

  float X[8], Y[8], Z[8], D[8];
#pragma unroll
  for (int i = 0; i < 8; ++i) {
    int p = i * 1024 + tid;
    X[i] = px[p * 3 + 0];
    Y[i] = px[p * 3 + 1];
    Z[i] = px[p * 3 + 2];
    D[i] = 1e10f;
  }

  __shared__ float sL[3];
  __shared__ float sBD[32];
  __shared__ int sBI[32];

  if (tid == 0) {
    newXyz[((size_t)b * S_ + 0) * 3 + 0] = px[0];
    newXyz[((size_t)b * S_ + 0) * 3 + 1] = px[1];
    newXyz[((size_t)b * S_ + 0) * 3 + 2] = px[2];
    sL[0] = px[0]; sL[1] = px[1]; sL[2] = px[2];
  }
  __syncthreads();

  for (int it = 1; it < S_; ++it) {
    float lx = sL[0], ly = sL[1], lz = sL[2];
    float bd = -1.f;
    int bi = 0x7fffffff;
#pragma unroll
    for (int i = 0; i < 8; ++i) {
      float dx = X[i] - lx, dy = Y[i] - ly, dz = Z[i] - lz;
      float d = dx * dx + dy * dy + dz * dz;
      D[i] = fminf(D[i], d);
      if (D[i] > bd) { bd = D[i]; bi = i * 1024 + tid; }
    }
#pragma unroll
    for (int off = 16; off > 0; off >>= 1) {
      float od = __shfl_down(bd, off, 32);
      int oi = __shfl_down(bi, off, 32);
      if (od > bd || (od == bd && oi < bi)) { bd = od; bi = oi; }
    }
    if (lane == 0) { sBD[wid] = bd; sBI[wid] = bi; }
    __syncthreads();
    if (tid == 0) {
      float mbd = sBD[0];
      int mbi = sBI[0];
      for (int w = 1; w < 32; ++w)
        if (sBD[w] > mbd || (sBD[w] == mbd && sBI[w] < mbi)) { mbd = sBD[w]; mbi = sBI[w]; }
      float ax = px[mbi * 3 + 0], ay = px[mbi * 3 + 1], az = px[mbi * 3 + 2];
      sL[0] = ax; sL[1] = ay; sL[2] = az;
      newXyz[((size_t)b * S_ + it) * 3 + 0] = ax;
      newXyz[((size_t)b * S_ + it) * 3 + 1] = ay;
      newXyz[((size_t)b * S_ + it) * 3 + 2] = az;
    }
    __syncthreads();
  }
}

// ---------------------------------------------------------------------------
__global__ void zero_kernel(float* __restrict__ p, int n) {
  int i = blockIdx.x * blockDim.x + threadIdx.x;
  if (i < n) p[i] = 0.f;
}

// Convert one weight matrix (cin,cout) f32 -> transposed padded f16 [cout][104]
__global__ void prep_w_kernel(const float* __restrict__ w, _Float16* __restrict__ dst,
                              int cout, int kreal) {
  int t = blockIdx.x * blockDim.x + threadIdx.x;
  if (t >= cout * WSTR) return;
  int n = t / WSTR, k = t % WSTR;
  dst[t] = (k < kreal) ? (_Float16)w[k * cout + n] : (_Float16)0.f;
}

// Convert points features f32 -> f16
__global__ void prep_p_kernel(const float* __restrict__ src, _Float16* __restrict__ dst,
                              int n) {
  int t = blockIdx.x * blockDim.x + threadIdx.x;
  if (t < n) dst[t] = (_Float16)src[t];
}

// ---------------------------------------------------------------------------
// Ball query: one wave per query point.
// ---------------------------------------------------------------------------
__global__ __launch_bounds__(256) void ball_kernel(const float* __restrict__ xyz,
                                                   const float* __restrict__ newXyz,
                                                   int* __restrict__ out,
                                                   float r2, int ns) {
  int g = blockIdx.x * 8 + (threadIdx.x >> 5);
  int lane = threadIdx.x & 31;
  if (g >= B_ * S_) return;
  int b = g / S_;
  const float* px = xyz + (size_t)b * N_ * 3;
  float qx = newXyz[(size_t)g * 3 + 0];
  float qy = newXyz[(size_t)g * 3 + 1];
  float qz = newXyz[(size_t)g * 3 + 2];
  int* o = out + (size_t)g * ns;

  int cnt = 0, first = -1;
  for (int base = 0; base < N_ && cnt < ns; base += 32) {
    int i = base + lane;
    float dx = px[i * 3 + 0] - qx;
    float dy = px[i * 3 + 1] - qy;
    float dz = px[i * 3 + 2] - qz;
    bool inb = (dx * dx + dy * dy + dz * dz) < r2;
    unsigned long long m64 = __ballot(inb);
    unsigned m = (unsigned)m64;
    if (first < 0 && m) first = base + (int)__ffsll((unsigned long long)m) - 1;
    int pos = cnt + __popc(m & ((1u << lane) - 1u));
    if (inb && pos < ns) o[pos] = i;
    cnt += __popc(m);
  }
  if (cnt < ns) {
    int f = (cnt > 0) ? first : 0;
    for (int p = cnt + lane; p < ns; p += 32) o[p] = f;
  }
}

// ---------------------------------------------------------------------------
// WMMA fragment loader (16x32 f16 A / 32x16 B from [row][K]-major LDS).
// ---------------------------------------------------------------------------
__device__ __forceinline__ v16h ldfrag(const _Float16* p0, int stride, int row,
                                       int kbase, int lane) {
  const _Float16* p = p0 + row * stride + kbase + ((lane & 16) ? 8 : 0);
  v8h lo = *(const v8h*)(p);
  v8h hi = *(const v8h*)(p + 16);
  union { v16h v; v8h h[2]; } u;
  u.h[0] = lo;
  u.h[1] = hi;
  return u.v;
}

// Compile-time-unrolled N-tile WMMA chain so the reuse_a hint is a constant:
// first WMMA of the chain loads A normally, subsequent ones set reuse_a=1.
template <int NTI, int NT>
struct WmmaRow {
  static __device__ __forceinline__ void run(v8f* acc, v16h a, const _Float16* wT,
                                             int kbase, int lane) {
    v16h bf = ldfrag(wT, WSTR, NTI * 16 + (lane & 15), kbase, lane);
    acc[NTI] = __builtin_amdgcn_wmma_f32_16x16x32_f16(false, a, false, bf,
                                                      (short)0, acc[NTI],
                                                      /*reuse_a=*/(NTI > 0),
                                                      /*reuse_b=*/false);
    WmmaRow<NTI + 1, NT>::run(acc, a, wT, kbase, lane);
  }
};
template <int NT>
struct WmmaRow<NT, NT> {
  static __device__ __forceinline__ void run(v8f*, v16h, const _Float16*, int, int) {}
};

// ---------------------------------------------------------------------------
// One fused 1x1-conv layer over a 128-row tile, 8 waves. A-fragment hoisted
// across the N-tile loop; all N-tile accumulators register-resident.
// TOUT: store output transposed [col][row] packed (one ds_store_b128 per lane
// per N-tile) for the max-pool consumer.
// ---------------------------------------------------------------------------
template <int KPAD, int COUT, bool TOUT>
__device__ __forceinline__ void mlp_layer(const _Float16* actIn, _Float16* actOut,
                                          _Float16* wT, float* sb,
                                          const _Float16* __restrict__ wsrc,
                                          const float* __restrict__ bias, int tid) {
  // straight f16 copy of preconverted transposed weights into LDS
  {
    const v8h* src = (const v8h*)wsrc;
    v8h* dst = (v8h*)wT;
    constexpr int n16 = COUT * WSTR / 8;
    for (int t = tid; t < n16; t += 256) dst[t] = src[t];
  }
  if (tid < COUT) sb[tid] = bias[tid];
  __syncthreads();

  const int wave = tid >> 5, lane = tid & 31;
  const int mrow = wave * 16 + (lane & 15);
  constexpr int NT = COUT / 16;

  v8f acc[NT];
  const v8f vzero = {0.f, 0.f, 0.f, 0.f, 0.f, 0.f, 0.f, 0.f};
#pragma unroll
  for (int nt = 0; nt < NT; ++nt) acc[nt] = vzero;

#pragma unroll
  for (int kt = 0; kt < KPAD / 32; ++kt) {
    v16h a = ldfrag(actIn, 136, mrow, kt * 32, lane);
    WmmaRow<0, NT>::run(acc, a, wT, kt * 32, lane);
  }

  const int r0 = wave * 16 + ((lane & 16) ? 8 : 0);
#pragma unroll
  for (int nt = 0; nt < NT; ++nt) {
    const int col = nt * 16 + (lane & 15);
    const float bv = sb[col];
    if constexpr (TOUT) {
      v8h h;
#pragma unroll
      for (int i = 0; i < 8; ++i) {
        float v = acc[nt][i] + bv;
        h[i] = (_Float16)(v > 0.f ? v : 0.f);
      }
      *(v8h*)(actOut + col * 136 + r0) = h;  // [col][row], contiguous rows
    } else {
#pragma unroll
      for (int i = 0; i < 8; ++i) {
        float v = acc[nt][i] + bv;
        v = v > 0.f ? v : 0.f;
        actOut[(r0 + i) * 136 + col] = (_Float16)v;  // [row][col]
      }
    }
  }
  __syncthreads();
}

// ---------------------------------------------------------------------------
// Fused gather + 3-layer MLP + neighborhood max. 128 grouped rows per block,
// 256 threads (8 wave32s). ~97KB LDS per workgroup (CDNA5 WGP: 320KB).
// ---------------------------------------------------------------------------
template <int NS, int C1, int C2, int C3>
__global__ __launch_bounds__(256) void mlp_kernel(
    const float* __restrict__ xyz, const _Float16* __restrict__ pointsH,
    const float* __restrict__ newXyz, const int* __restrict__ ballIdx,
    const _Float16* __restrict__ w0, const float* __restrict__ b0,
    const _Float16* __restrict__ w1, const float* __restrict__ b1,
    const _Float16* __restrict__ w2, const float* __restrict__ b2,
    float* __restrict__ feats, int chanOff) {
  __shared__ __align__(16) _Float16 actA[128 * 136];
  __shared__ __align__(16) _Float16 actB[128 * 136];
  __shared__ __align__(16) _Float16 wT[128 * WSTR];
  __shared__ float sb[128];

  const int tid = threadIdx.x;
  const int row0 = blockIdx.x * 128;

  // Build grouped rows: [pointsH[idx] (32ch f16) | xyz[idx]-new_xyz (3) | 0 pad]
  {
    int row = tid & 127, half = tid >> 7;
    int grow = row0 + row;
    int b = grow / (S_ * NS);
    int rem = grow % (S_ * NS);
    int s = rem / NS, j = rem % NS;
    int idx = ballIdx[(size_t)(b * S_ + s) * NS + j];
    _Float16* d = actA + row * 136;
    if (half == 0) {
      const v8h* pf = (const v8h*)(pointsH + ((size_t)b * N_ + idx) * C_);
#pragma unroll
      for (int t = 0; t < 4; ++t) ((v8h*)d)[t] = pf[t];
    } else {
      const float* pp = xyz + ((size_t)b * N_ + idx) * 3;
      const float* q = newXyz + (size_t)(b * S_ + s) * 3;
      d[32] = (_Float16)(pp[0] - q[0]);
      d[33] = (_Float16)(pp[1] - q[1]);
      d[34] = (_Float16)(pp[2] - q[2]);
#pragma unroll
      for (int k = 35; k < 64; ++k) d[k] = (_Float16)0.f;
    }
  }

  mlp_layer<64, C1, false>(actA, actB, wT, sb, w0, b0, tid);
  mlp_layer<C1, C2, false>(actB, actA, wT, sb, w1, b1, tid);
  mlp_layer<C2, C3, true>(actA, actB, wT, sb, w2, b2, tid);  // transposed out

  // Max over neighborhood; final buffer is [col][row] so rows are contiguous.
  constexpr int RQ = NS;        // NS <= 128 == tile rows
  constexpr int QT = 128 / RQ;
  for (int t = tid; t < QT * C3; t += 256) {
    int qi = t / C3, c = t % C3;
    const _Float16* src = actB + c * 136 + qi * RQ;
    float m = 0.f;
#pragma unroll 4
    for (int r = 0; r < RQ; ++r) m = fmaxf(m, (float)src[r]);
    int g0 = row0 + qi * RQ;
    int b = g0 / (S_ * NS);
    int s = (g0 % (S_ * NS)) / NS;
    atomicMax((unsigned int*)&feats[(size_t)(b * S_ + s) * FTOT + chanOff + c],
              __float_as_uint(m));
  }
}

// ---------------------------------------------------------------------------
extern "C" void kernel_launch(void* const* d_in, const int* in_sizes, int n_in,
                              void* d_out, int out_size, void* d_ws, size_t ws_size,
                              hipStream_t stream) {
  const float* xyz = (const float*)d_in[0];
  const float* points = (const float*)d_in[1];

  float* newXyz = (float*)d_out;                       // (B,S,3)
  float* feats = (float*)d_out + (size_t)B_ * S_ * 3;  // (B,S,320)

  // workspace layout
  int* ball0 = (int*)d_ws;
  int* ball1 = ball0 + (size_t)B_ * S_ * 16;
  int* ball2 = ball1 + (size_t)B_ * S_ * 32;
  _Float16* ptsH = (_Float16*)(ball2 + (size_t)B_ * S_ * 128);
  _Float16* wH = ptsH + (size_t)B_ * N_ * C_;

  // per-matrix f16 transposed-weight slots, [cout][104]
  const int couts[9]  = {32, 32, 64, 64, 64, 128, 64, 96, 128};
  const int kreals[9] = {35, 32, 32, 35, 64, 64,  35, 64, 96};
  _Float16* wptr[9];
  {
    _Float16* p = wH;
    for (int i = 0; i < 9; ++i) { wptr[i] = p; p += (size_t)couts[i] * WSTR; }
  }

  fps_kernel<<<B_, 1024, 0, stream>>>(xyz, newXyz);

  const int nf = B_ * S_ * FTOT;
  zero_kernel<<<(nf + 255) / 256, 256, 0, stream>>>(feats, nf);

  // preconvert weights + points to f16
  for (int i = 0; i < 9; ++i) {
    const float* w = (const float*)d_in[2 + 2 * i];
    int total = couts[i] * WSTR;
    prep_w_kernel<<<(total + 255) / 256, 256, 0, stream>>>(w, wptr[i], couts[i], kreals[i]);
  }
  {
    int np = B_ * N_ * C_;
    prep_p_kernel<<<(np + 255) / 256, 256, 0, stream>>>(points, ptsH, np);
  }

  ball_kernel<<<B_ * S_ / 8, 256, 0, stream>>>(xyz, newXyz, ball0, 0.1f * 0.1f, 16);
  ball_kernel<<<B_ * S_ / 8, 256, 0, stream>>>(xyz, newXyz, ball1, 0.2f * 0.2f, 32);
  ball_kernel<<<B_ * S_ / 8, 256, 0, stream>>>(xyz, newXyz, ball2, 0.4f * 0.4f, 128);

  const float* b0_0 = (const float*)d_in[3];
  const float* b0_1 = (const float*)d_in[5];
  const float* b0_2 = (const float*)d_in[7];
  const float* b1_0 = (const float*)d_in[9];
  const float* b1_1 = (const float*)d_in[11];
  const float* b1_2 = (const float*)d_in[13];
  const float* b2_0 = (const float*)d_in[15];
  const float* b2_1 = (const float*)d_in[17];
  const float* b2_2 = (const float*)d_in[19];

  mlp_kernel<16, 32, 32, 64><<<B_ * S_ * 16 / 128, 256, 0, stream>>>(
      xyz, ptsH, newXyz, ball0, wptr[0], b0_0, wptr[1], b0_1, wptr[2], b0_2, feats, 0);
  mlp_kernel<32, 64, 64, 128><<<B_ * S_ * 32 / 128, 256, 0, stream>>>(
      xyz, ptsH, newXyz, ball1, wptr[3], b1_0, wptr[4], b1_1, wptr[5], b1_2, feats, 64);
  mlp_kernel<128, 64, 96, 128><<<B_ * S_ * 128 / 128, 256, 0, stream>>>(
      xyz, ptsH, newXyz, ball2, wptr[6], b2_0, wptr[7], b2_1, wptr[8], b2_2, feats, 192);
}
